// Actor_43173011259890
// MI455X (gfx1250) — matrix operand order repaired and checked by reading, not compiled, not run on hardware
//
#include <hip/hip_runtime.h>
#include <hip/hip_bf16.h>
#include <math.h>

// ---------------- types for WMMA (gfx1250, wave32) ----------------
typedef __bf16 bf16_t;
typedef __attribute__((ext_vector_type(8)))  bf16_t v8bf;
typedef __attribute__((ext_vector_type(16))) bf16_t v16bf;
typedef __attribute__((ext_vector_type(8)))  float  v8f;

#define N_NODES 2048
#define T_STEPS 14
#define K_DELAY 3
#define H_DIM   128
#define IN_DIM  6
#define OBS_DIM 6
#define GAM_DIM 2
#define D_DIM   136   // H + OBS + GAMMA
#define D_PAD   160   // D padded to multiple of 32 for WMMA K
#define G_MAP   512   // 4*H
#define G_LSTM  544   // 4*D

// ---------------------------------------------------------------------------
// rowsum of A (per row) -> inv_rowsum  (wave-per-row, coalesced lane-strided)
// ---------------------------------------------------------------------------
__global__ void row_inv_kernel(const int* __restrict__ A, float* __restrict__ inv_rowsum, int N) {
    int wid  = (blockIdx.x * blockDim.x + threadIdx.x) >> 5;
    int lane = threadIdx.x & 31;
    if (wid >= N) return;
    const int* row = A + (size_t)wid * N;
    int s = 0;
    for (int j = lane; j < N; j += 32) s += row[j];
    #pragma unroll
    for (int off = 16; off; off >>= 1) s += __shfl_xor(s, off, 32);
    if (lane == 0) inv_rowsum[wid] = (s > 0) ? (1.0f / (float)s) : 0.0f;
}

// ---------------------------------------------------------------------------
// column pass: deg (col sums), aggr = (A^T @ x) / max(deg,1), and A -> bf16
// ---------------------------------------------------------------------------
__global__ void col_aggr_cvt_kernel(const int* __restrict__ A, const float* __restrict__ x,
                                    bf16_t* __restrict__ Abf, float* __restrict__ aggr, int N) {
    int j = blockIdx.x * blockDim.x + threadIdx.x;
    if (j >= N) return;
    int d = 0;
    float acc[IN_DIM];
    #pragma unroll
    for (int c = 0; c < IN_DIM; c++) acc[c] = 0.0f;
    for (int i = 0; i < N; i++) {
        int v  = A[(size_t)i * N + j];
        float f = (float)v;
        Abf[(size_t)i * N + j] = (bf16_t)f;
        d += (v != 0);
        const float* xr = x + (size_t)i * IN_DIM;
        #pragma unroll
        for (int c = 0; c < IN_DIM; c++) acc[c] += f * xr[c];
    }
    float id = 1.0f / fmaxf((float)d, 1.0f);
    #pragma unroll
    for (int c = 0; c < IN_DIM; c++) aggr[(size_t)j * IN_DIM + c] = acc[c] * id;
}

// ---------------------------------------------------------------------------
// SAGE: out = aggr@wl^T + bl + x@wr^T ; row-normalize ; relu  -> xi (f32+bf16)
// ---------------------------------------------------------------------------
__global__ void sage_xi_kernel(const float* __restrict__ aggr, const float* __restrict__ x,
                               const float* __restrict__ wl, const float* __restrict__ bl,
                               const float* __restrict__ wr,
                               float* __restrict__ xi_f32, bf16_t* __restrict__ xi_bf) {
    int n = blockIdx.x;
    int h = threadIdx.x;   // 0..127
    __shared__ float ar[IN_DIM], xr[IN_DIM];
    __shared__ float red[H_DIM];
    if (h < IN_DIM) {
        ar[h] = aggr[(size_t)n * IN_DIM + h];
        xr[h] = x[(size_t)n * IN_DIM + h];
    }
    __syncthreads();
    float o = bl[h];
    #pragma unroll
    for (int c = 0; c < IN_DIM; c++)
        o += ar[c] * wl[h * IN_DIM + c] + xr[c] * wr[h * IN_DIM + c];
    red[h] = o * o;
    __syncthreads();
    for (int s = 64; s; s >>= 1) {
        if (h < s) red[h] += red[h + s];
        __syncthreads();
    }
    float nrm = sqrtf(red[0]);
    float v = fmaxf(o / fmaxf(nrm, 1e-12f), 0.0f);
    xi_f32[(size_t)n * H_DIM + h] = v;
    xi_bf [(size_t)n * H_DIM + h] = (bf16_t)v;
}

// ---------------------------------------------------------------------------
// Bst[k][h][m] = bf16( delayed_prev[k+1][m][h] * inv_rowsum[m] )   (k=0,1)
// ---------------------------------------------------------------------------
__global__ void scale_bst_kernel(const float* __restrict__ delayed_prev,
                                 const float* __restrict__ inv_rowsum,
                                 bf16_t* __restrict__ Bst) {
    int idx = blockIdx.x * blockDim.x + threadIdx.x;   // 2*N*H total
    if (idx >= 2 * N_NODES * H_DIM) return;
    int k = idx / (N_NODES * H_DIM);
    int r = idx % (N_NODES * H_DIM);
    int m = r / H_DIM;
    int h = r % H_DIM;
    float v = delayed_prev[(size_t)(k + 1) * N_NODES * H_DIM + (size_t)m * H_DIM + h] * inv_rowsum[m];
    Bst[(size_t)k * H_DIM * N_NODES + (size_t)h * N_NODES + m] = (bf16_t)v;
}

// ---------------------------------------------------------------------------
// Generic dual-phase WMMA GEMM (NT), MS x GS sub-tiles of 16x16 per wave:
//   C[M,G] = A1[M,K1] @ W1[G,K1]^T (+ A2[M,K2] @ W2[G,K2]^T) (+ bias[G])
// blockIdx.y batches W1/C (A shared across batch).
// Fragments use contiguous k = 16*hsel + s for both A and W (see round-1 notes).
// ---------------------------------------------------------------------------
template<int MS, int GS>
__device__ __forceinline__ void wmma_phase(v8f acc[MS][GS], const bf16_t* __restrict__ A,
                                           const bf16_t* __restrict__ W, int K,
                                           int m0, int g0, int lr, int hsel) {
    const bf16_t* ar[MS];
    const bf16_t* wr[GS];
    #pragma unroll
    for (int i = 0; i < MS; i++) ar[i] = A + (size_t)(m0 + i * 16 + lr) * K + hsel * 16;
    #pragma unroll
    for (int j = 0; j < GS; j++) wr[j] = W + (size_t)(g0 + j * 16 + lr) * K + hsel * 16;
    for (int k = 0; k < K; k += 32) {
        __builtin_prefetch(ar[0] + k + 128, 0, 1);
        __builtin_prefetch(wr[0] + k + 128, 0, 1);
        v16bf av[MS], wv[GS];
        #pragma unroll
        for (int i = 0; i < MS; i++) {
            v8bf alo = *(const v8bf*)(ar[i] + k);
            v8bf ahi = *(const v8bf*)(ar[i] + k + 8);
            #pragma unroll
            for (int e = 0; e < 8; e++) { av[i][e] = alo[e]; av[i][e + 8] = ahi[e]; }
        }
        #pragma unroll
        for (int j = 0; j < GS; j++) wv[j] = *(const v16bf*)(wr[j] + k);
        #pragma unroll
        for (int i = 0; i < MS; i++)
            #pragma unroll
            for (int j = 0; j < GS; j++)
                acc[i][j] = __builtin_amdgcn_wmma_f32_16x16x32_bf16(
                    false, av[i], false, wv[j], (short)0, acc[i][j], false, false);
    }
}

template<int MS, int GS>
__global__ __launch_bounds__(256)
void gemm_wmma_nt(float* __restrict__ C, bf16_t* __restrict__ Cbf,
                  const bf16_t* __restrict__ A1, const bf16_t* __restrict__ W1, int K1,
                  const bf16_t* __restrict__ A2, const bf16_t* __restrict__ W2, int K2,
                  const float* __restrict__ bias, int M, int G,
                  long w_bstride, long c_bstride) {
    // batch (merged GEMM: k=0,1 share A, different W / C)
    int b = blockIdx.y;
    W1 += (size_t)b * w_bstride;
    C  += (size_t)b * c_bstride;
    if (Cbf) Cbf += (size_t)b * c_bstride;

    int wave   = (blockIdx.x * blockDim.x + threadIdx.x) >> 5;
    int tilesG = G / (16 * GS);
    int tg = wave % tilesG;
    int tm = wave / tilesG;
    if (tm >= M / (16 * MS)) return;     // wave-uniform
    int lane = threadIdx.x & 31;
    int lr   = lane & 15;
    int hsel = lane >> 4;
    int m0 = tm * 16 * MS;
    int g0 = tg * 16 * GS;

    // --- CDNA5 async-LDS demo: fire-and-forget 16B/lane stage of A into LDS ---
    __shared__ bf16_t astage[256];
    {
        unsigned lds = (unsigned)(size_t)(&astage[0]) + (unsigned)(lane * 16);
        const bf16_t* gsrc = A1 + (size_t)lane * 8;
        asm volatile("global_load_async_to_lds_b128 %0, %1, off" :: "v"(lds), "v"(gsrc) : "memory");
        asm volatile("s_wait_asynccnt 0x0" ::: "memory");
    }

    v8f acc[MS][GS];
    #pragma unroll
    for (int i = 0; i < MS; i++)
        #pragma unroll
        for (int j = 0; j < GS; j++)
            acc[i][j] = (v8f){0.f, 0.f, 0.f, 0.f, 0.f, 0.f, 0.f, 0.f};

    wmma_phase<MS, GS>(acc, A1, W1, K1, m0, g0, lr, hsel);
    if (A2) wmma_phase<MS, GS>(acc, A2, W2, K2, m0, g0, lr, hsel);

    #pragma unroll
    for (int j = 0; j < GS; j++) {
        int col = g0 + j * 16 + lr;
        float bv = bias ? bias[col] : 0.0f;
        #pragma unroll
        for (int i = 0; i < MS; i++) {
            #pragma unroll
            for (int r = 0; r < 8; r++) {
                int row = m0 + i * 16 + r + hsel * 8;
                float v = acc[i][j][r] + bv;
                C[(size_t)row * G + col] = v;
                if (Cbf) Cbf[(size_t)row * G + col] = (bf16_t)v;
            }
        }
    }
}

// ---------------------------------------------------------------------------
// LSTM pointwise: gates z = [i|f|g|o] (f32, stride 4*Hd), update c, emit h
// ---------------------------------------------------------------------------
__global__ void lstm_pointwise_kernel(const float* __restrict__ z, float* __restrict__ c,
                                      bf16_t* __restrict__ hbf, float* __restrict__ hf32,
                                      int B, int Hd, int Hpad) {
    int idx = blockIdx.x * blockDim.x + threadIdx.x;
    if (idx >= B * Hd) return;
    int n  = idx / Hd;
    int hd = idx % Hd;
    int G  = 4 * Hd;
    const float* zr = z + (size_t)n * G;
    float zi = zr[hd], zf = zr[Hd + hd], zg = zr[2 * Hd + hd], zo = zr[3 * Hd + hd];
    float si = 1.0f / (1.0f + expf(-zi));
    float sf = 1.0f / (1.0f + expf(-zf));
    float so = 1.0f / (1.0f + expf(-zo));
    float c2 = sf * c[idx] + si * tanhf(zg);
    float h2 = so * tanhf(c2);
    c[idx] = c2;
    hbf[(size_t)n * Hpad + hd] = (bf16_t)h2;
    if (hf32) hf32[idx] = h2;
}

// feat tail: cols 128..159 of padded bf16 feature row = [obs(6) | gamma(2) | zeros]
__global__ void feat_tail_kernel(const float* __restrict__ obs, const float* __restrict__ gamma,
                                 bf16_t* __restrict__ feat_t) {
    int idx = blockIdx.x * blockDim.x + threadIdx.x;   // N*32
    if (idx >= N_NODES * 32) return;
    int n = idx >> 5, j = idx & 31;
    float v = 0.0f;
    if (j < OBS_DIM)                v = obs[(size_t)n * OBS_DIM + j];
    else if (j < OBS_DIM + GAM_DIM) v = gamma[(size_t)n * GAM_DIM + (j - OBS_DIM)];
    feat_t[(size_t)n * D_PAD + H_DIM + j] = (bf16_t)v;
}

// head: out = relu(h @ lin_w^T + lin_b) @ lin1_w^T + lin1_b   (block per node)
__global__ void head_kernel(const float* __restrict__ hfin,
                            const float* __restrict__ lin_w, const float* __restrict__ lin_b,
                            const float* __restrict__ lin1_w, const float* __restrict__ lin1_b,
                            float* __restrict__ out) {
    int n = blockIdx.x, tid = threadIdx.x;     // blockDim = 160
    __shared__ float hr[D_DIM];
    __shared__ float h2[D_DIM];
    if (tid < D_DIM) hr[tid] = hfin[(size_t)n * D_DIM + tid];
    __syncthreads();
    if (tid < D_DIM) {
        float s = lin_b[tid];
        for (int c = 0; c < D_DIM; c++) s += hr[c] * lin_w[tid * D_DIM + c];
        h2[tid] = fmaxf(s, 0.0f);
    }
    __syncthreads();
    if (tid < 2) {
        float s = lin1_b[tid];
        for (int c = 0; c < D_DIM; c++) s += h2[c] * lin1_w[tid * D_DIM + c];
        out[(size_t)n * 2 + tid] = s;
    }
}

// f32 [G,K] -> bf16 [G,Kp] with zero pad
__global__ void cvt_pad_kernel(bf16_t* __restrict__ dst, const float* __restrict__ src,
                               int G, int K, int Kp) {
    int idx = blockIdx.x * blockDim.x + threadIdx.x;
    if (idx >= G * Kp) return;
    int g = idx / Kp, k = idx % Kp;
    dst[idx] = (bf16_t)(k < K ? src[(size_t)g * K + k] : 0.0f);
}

__global__ void bias_sum_kernel(float* __restrict__ dst, const float* __restrict__ a,
                                const float* __restrict__ b, int n) {
    int idx = blockIdx.x * blockDim.x + threadIdx.x;
    if (idx < n) dst[idx] = a[idx] + b[idx];
}

// ---------------------------------------------------------------------------
extern "C" void kernel_launch(void* const* d_in, const int* in_sizes, int n_in,
                              void* d_out, int out_size, void* d_ws, size_t ws_size,
                              hipStream_t stream) {
    const float* x_queue    = (const float*)d_in[1];
    const int*   a_queue    = (const int*)  d_in[2];
    const float* obs_queue  = (const float*)d_in[3];
    const float* gam_queue  = (const float*)d_in[5];
    const float* sage_wl    = (const float*)d_in[6];
    const float* sage_bl    = (const float*)d_in[7];
    const float* sage_wr    = (const float*)d_in[8];
    const float* map_wih    = (const float*)d_in[9];
    const float* map_whh    = (const float*)d_in[10];
    const float* map_bih    = (const float*)d_in[11];
    const float* map_bhh    = (const float*)d_in[12];
    const float* lstm_wih   = (const float*)d_in[13];
    const float* lstm_whh   = (const float*)d_in[14];
    const float* lstm_bih   = (const float*)d_in[15];
    const float* lstm_bhh   = (const float*)d_in[16];
    const float* lin_w      = (const float*)d_in[17];
    const float* lin_b      = (const float*)d_in[18];
    const float* lin1_w     = (const float*)d_in[19];
    const float* lin1_b     = (const float*)d_in[20];
    float* out = (float*)d_out;

    const int N = N_NODES;
    char* ws = (char*)d_ws;
    size_t off = 0;
    auto alloc = [&](size_t bytes) -> void* {
        off = (off + 255) & ~(size_t)255;
        void* p = ws + off;
        off += bytes;
        return p;
    };

    bf16_t* Abf        = (bf16_t*)alloc((size_t)N * N * 2);
    float*  inv_rowsum = (float*) alloc((size_t)N * 4);
    float*  aggr       = (float*) alloc((size_t)N * IN_DIM * 4);
    float*  delayedA   = (float*) alloc((size_t)K_DELAY * N * H_DIM * 4);
    float*  delayedB   = (float*) alloc((size_t)K_DELAY * N * H_DIM * 4);
    bf16_t* xbf        = (bf16_t*)alloc((size_t)K_DELAY * N * H_DIM * 2);
    bf16_t* Bst        = (bf16_t*)alloc((size_t)2 * H_DIM * N * 2);
    bf16_t* map_hbf    = (bf16_t*)alloc((size_t)N * H_DIM * 2);
    float*  map_c      = (float*) alloc((size_t)N * H_DIM * 4);
    float*  zbuf       = (float*) alloc((size_t)N * G_LSTM * 4);
    bf16_t* featbf     = (bf16_t*)alloc((size_t)T_STEPS * N * D_PAD * 2);
    bf16_t* lstm_hbf   = (bf16_t*)alloc((size_t)N * D_PAD * 2);
    float*  lstm_h32   = (float*) alloc((size_t)N * D_DIM * 4);
    float*  lstm_c     = (float*) alloc((size_t)N * D_DIM * 4);
    bf16_t* wmapih     = (bf16_t*)alloc((size_t)G_MAP * H_DIM * 2);
    bf16_t* wmaphh     = (bf16_t*)alloc((size_t)G_MAP * H_DIM * 2);
    bf16_t* wlstmih    = (bf16_t*)alloc((size_t)G_LSTM * D_PAD * 2);
    bf16_t* wlstmhh    = (bf16_t*)alloc((size_t)G_LSTM * D_PAD * 2);
    float*  bias_map   = (float*) alloc((size_t)G_MAP * 4);
    float*  bias_lstm  = (float*) alloc((size_t)G_LSTM * 4);
    if (off > ws_size) return;   // workspace too small: bail deterministically

    // ---- weight conversion (per call; cheap) ----
    cvt_pad_kernel<<<(G_MAP * H_DIM + 255) / 256, 256, 0, stream>>>(wmapih, map_wih, G_MAP, H_DIM, H_DIM);
    cvt_pad_kernel<<<(G_MAP * H_DIM + 255) / 256, 256, 0, stream>>>(wmaphh, map_whh, G_MAP, H_DIM, H_DIM);
    cvt_pad_kernel<<<(G_LSTM * D_PAD + 255) / 256, 256, 0, stream>>>(wlstmih, lstm_wih, G_LSTM, D_DIM, D_PAD);
    cvt_pad_kernel<<<(G_LSTM * D_PAD + 255) / 256, 256, 0, stream>>>(wlstmhh, lstm_whh, G_LSTM, D_DIM, D_PAD);
    bias_sum_kernel<<<(G_MAP + 255) / 256, 256, 0, stream>>>(bias_map, map_bih, map_bhh, G_MAP);
    bias_sum_kernel<<<(G_LSTM + 255) / 256, 256, 0, stream>>>(bias_lstm, lstm_bih, lstm_bhh, G_LSTM);

    hipMemsetAsync(delayedA, 0, (size_t)K_DELAY * N * H_DIM * 4, stream);

    float* dprev = delayedA;
    float* dcur  = delayedB;

    for (int t = 0; t < T_STEPS; t++) {
        const int*   At   = a_queue   + (size_t)t * N * N;
        const float* xt   = x_queue   + (size_t)t * N * IN_DIM;
        const float* obst = obs_queue + (size_t)t * N * OBS_DIM;
        const float* gamt = gam_queue + (size_t)t * N * GAM_DIM;

        row_inv_kernel<<<N / 8, 256, 0, stream>>>(At, inv_rowsum, N);
        col_aggr_cvt_kernel<<<N / 256, 256, 0, stream>>>(At, xt, Abf, aggr, N);
        sage_xi_kernel<<<N, H_DIM, 0, stream>>>(aggr, xt, sage_wl, sage_bl, sage_wr,
                                                dcur + (size_t)2 * N * H_DIM,
                                                xbf  + (size_t)2 * N * H_DIM);
        scale_bst_kernel<<<(2 * N * H_DIM) / 256, 256, 0, stream>>>(dprev, inv_rowsum, Bst);

        // merged[k] = A @ (inv_rowsum * delayed_prev[k+1]), k=0,1 in one batched launch
        {
            int tiles = (N / 16) * (H_DIM / 32);       // MS=1, GS=2
            dim3 grid(tiles / 8, 2);
            gemm_wmma_nt<1, 2><<<grid, 256, 0, stream>>>(
                dcur, xbf, Abf, Bst, N,
                nullptr, nullptr, 0, nullptr, N, H_DIM,
                (long)H_DIM * N /*W batch stride*/, (long)N * H_DIM /*C batch stride*/);
        }

        // ---- map LSTM over temp[0..2], zero init state ----
        hipMemsetAsync(map_hbf, 0, (size_t)N * H_DIM * 2, stream);
        hipMemsetAsync(map_c,   0, (size_t)N * H_DIM * 4, stream);
        for (int s = 0; s < K_DELAY; s++) {
            int tiles = (N / 32) * (G_MAP / 32);       // MS=2, GS=2
            gemm_wmma_nt<2, 2><<<tiles / 8, 256, 0, stream>>>(
                zbuf, nullptr,
                xbf + (size_t)s * N * H_DIM, wmapih, H_DIM,
                map_hbf, wmaphh, H_DIM,
                bias_map, N, G_MAP, 0, 0);
            bf16_t* hdst = (s == K_DELAY - 1) ? (featbf + (size_t)t * N * D_PAD) : map_hbf;
            int hpad     = (s == K_DELAY - 1) ? D_PAD : H_DIM;
            lstm_pointwise_kernel<<<(N * H_DIM) / 256, 256, 0, stream>>>(
                zbuf, map_c, hdst, nullptr, N, H_DIM, hpad);
        }
        feat_tail_kernel<<<(N * 32) / 256, 256, 0, stream>>>(obst, gamt,
                                                             featbf + (size_t)t * N * D_PAD);
        float* tmp = dprev; dprev = dcur; dcur = tmp;
    }

    // ---- final LSTM over feats[2..13] ----
    hipMemsetAsync(lstm_hbf, 0, (size_t)N * D_PAD * 2, stream);
    hipMemsetAsync(lstm_c,   0, (size_t)N * D_DIM * 4, stream);
    for (int s = 0; s < 12; s++) {
        int t = K_DELAY - 1 + s;
        int tiles = (N / 32) * (G_LSTM / 32);          // MS=2, GS=2
        gemm_wmma_nt<2, 2><<<tiles / 8, 256, 0, stream>>>(
            zbuf, nullptr,
            featbf + (size_t)t * N * D_PAD, wlstmih, D_PAD,
            lstm_hbf, wlstmhh, D_PAD,
            bias_lstm, N, G_LSTM, 0, 0);
        lstm_pointwise_kernel<<<(N * D_DIM + 255) / 256, 256, 0, stream>>>(
            zbuf, lstm_c, lstm_hbf, lstm_h32, N, D_DIM, D_PAD);
    }

    // ---- head ----
    head_kernel<<<N, 160, 0, stream>>>(lstm_h32, lin_w, lin_b, lin1_w, lin1_b, out);
}
